// MultiHeadAttention_73770358276773
// MI455X (gfx1250) — matrix operand order repaired and checked by reading, not compiled, not run on hardware
//
#include <hip/hip_runtime.h>

// MI455X / gfx1250, wave32. bf16 WMMA (v_wmma_f32_16x16x32_bf16) GEMM pipeline
// for multi-head attention. Double-buffered LDS, 64-wide K stages (2 WMMA
// sub-steps per barrier), branch-free templated tile loaders, LDS-cached
// softmax. Scratch in d_ws (~64MB); attn written directly into d_out.

#define BB 4
#define TT 2048
#define DD 1024
#define HH 16
#define DH 64
#define MTOT (BB * TT)  // 8192

typedef __attribute__((ext_vector_type(16))) __bf16 v16bf;
typedef __attribute__((ext_vector_type(8)))  __bf16 v8bf;
typedef __attribute__((ext_vector_type(8)))  float  v8f;

#define WMMA_BF16(a, b, c) \
    __builtin_amdgcn_wmma_f32_16x16x32_bf16(false, (a), false, (b), (short)0, (c), false, false)

// ---------------------------------------------------------------------------
// Fragment loads from LDS (wave32 layouts, CDNA5 ISA 7.12.2). A sub-tile in
// LDS is [row][32] with K-groups permuted [k0-7|k16-23|k8-15|k24-31], so lane
// (half = lane>>4, m = lane&15) reads 16 contiguous bf16 (one 32B LDS read).
static __device__ __forceinline__ v16bf ldsA_frag(const __bf16* t, int rowBase) {
    const int lane = threadIdx.x & 31;
    return *(const v16bf*)(t + (size_t)(rowBase + (lane & 15)) * 32 + (lane >> 4) * 16);
}
// B sub-tile in LDS: [n][32], same K-group permutation as A; lane needs K=8h+0..7,16+8h+0..7 (h=lane>>4)
// at N = lane&15 -> one contiguous v16bf.
static __device__ __forceinline__ v16bf ldsB_frag(const __bf16* t, int nBase) {
    const int lane = threadIdx.x & 31;
    return *(const v16bf*)(t + (size_t)(nBase + (lane & 15)) * 32 + (lane >> 4) * 16);
}

// ---------------------------------------------------------------------------
// Branch-free cooperative tile loaders (256 threads, compile-time shapes).
// LDS layout: [KB/32 sub-tiles][ROWS][32]. Each chunk = 16 source elements.
template <int ROWS, int KB>
static __device__ __forceinline__ void loadA_f32(__bf16* lds, const float* g, int ldg) {
    constexpr int CPR = KB / 16;
    constexpr int CHUNKS = ROWS * CPR;
#pragma unroll
    for (int c0 = 0; c0 < CHUNKS; c0 += 256) {
        const int c = c0 + (int)threadIdx.x;
        if (CHUNKS % 256 == 0 || c < CHUNKS) {
            const int row = c / CPR;
            const int k16 = (c % CPR) * 16;
            const int sub = k16 >> 5;
            const int h   = (k16 >> 4) & 1;
            const float* s = g + (size_t)row * ldg + k16;
            __bf16* base = lds + ((size_t)sub * ROWS + row) * 32;
            v8bf p0, p1;
#pragma unroll
            for (int i = 0; i < 8; ++i) {
                p0[i] = (__bf16)s[i];
                p1[i] = (__bf16)s[8 + i];
            }
            *(v8bf*)(base + h * 8)      = p0;  // k(h*16 .. +7)
            *(v8bf*)(base + h * 8 + 16) = p1;  // k(h*16+8 .. +15)
        }
    }
}
template <int ROWS, int KB>
static __device__ __forceinline__ void loadA_bf16(__bf16* lds, const __bf16* g, int ldg) {
    constexpr int CPR = KB / 16;
    constexpr int CHUNKS = ROWS * CPR;
#pragma unroll
    for (int c0 = 0; c0 < CHUNKS; c0 += 256) {
        const int c = c0 + (int)threadIdx.x;
        if (CHUNKS % 256 == 0 || c < CHUNKS) {
            const int row = c / CPR;
            const int k16 = (c % CPR) * 16;
            const int sub = k16 >> 5;
            const int h   = (k16 >> 4) & 1;
            const __bf16* s = g + (size_t)row * ldg + k16;
            __bf16* base = lds + ((size_t)sub * ROWS + row) * 32;
            *(v8bf*)(base + h * 8)      = *(const v8bf*)(s);
            *(v8bf*)(base + h * 8 + 16) = *(const v8bf*)(s + 8);
        }
    }
}
template <int ROWS, int KB>
static __device__ __forceinline__ void loadB_f32(__bf16* lds, const float* g, int ldg) {
    constexpr int CPR = KB / 16;
    constexpr int CHUNKS = ROWS * CPR;
#pragma unroll
    for (int c0 = 0; c0 < CHUNKS; c0 += 256) {
        const int c = c0 + (int)threadIdx.x;
        if (CHUNKS % 256 == 0 || c < CHUNKS) {
            const int row = c / CPR;
            const int k16 = (c % CPR) * 16;
            const int sub = k16 >> 5;
            const int h   = (k16 >> 4) & 1;
            const float* s = g + (size_t)row * ldg + k16;
            __bf16* base = lds + ((size_t)sub * ROWS + row) * 32;
            v8bf p0, p1;
#pragma unroll
            for (int i = 0; i < 8; ++i) {
                p0[i] = (__bf16)s[i];
                p1[i] = (__bf16)s[8 + i];
            }
            *(v8bf*)(base + h * 8)      = p0;
            *(v8bf*)(base + h * 8 + 16) = p1;
        }
    }
}
template <int ROWS, int KB>
static __device__ __forceinline__ void loadB_bf16(__bf16* lds, const __bf16* g, int ldg) {
    constexpr int CPR = KB / 16;
    constexpr int CHUNKS = ROWS * CPR;
#pragma unroll
    for (int c0 = 0; c0 < CHUNKS; c0 += 256) {
        const int c = c0 + (int)threadIdx.x;
        if (CHUNKS % 256 == 0 || c < CHUNKS) {
            const int row = c / CPR;
            const int k16 = (c % CPR) * 16;
            const int sub = k16 >> 5;
            const int h   = (k16 >> 4) & 1;
            __bf16* base = lds + ((size_t)sub * ROWS + row) * 32;
            *(v8bf*)(base + h * 8)      = *(const v8bf*)(g + (size_t)row * ldg + k16);
            *(v8bf*)(base + h * 8 + 16) = *(const v8bf*)(g + (size_t)row * ldg + k16 + 8);
        }
    }
}

// ---------------------------------------------------------------------------
// Kernel 1: QKV projections. out = x @ W^T + b (NT GEMM: A=[m][k], B=W=[n][k]).
// z = 0/1/2 -> Q/K/V. Q scaled by 1/sqrt(DH)=0.125. Q,K -> [b,h,t,d] bf16;
// V -> transposed [b,h,d,t] bf16 for the attn@V GEMM. 128x128 tile, KB=64,
// double-buffered LDS, one barrier per stage, 16 wmma per barrier.
__global__ __launch_bounds__(256) void k_qkv(
    const float* __restrict__ x,
    const float* __restrict__ wq, const float* __restrict__ bq,
    const float* __restrict__ wk, const float* __restrict__ bk,
    const float* __restrict__ wv, const float* __restrict__ bv,
    __bf16* __restrict__ qw, __bf16* __restrict__ kw, __bf16* __restrict__ vtw) {
    __shared__ __align__(32) __bf16 lA[2][2 * 128 * 32];  // 16KB x2
    __shared__ __align__(32) __bf16 lB[2][2 * 128 * 32];  // 16KB x2
    const int which = blockIdx.z;
    const float* w    = (which == 0) ? wq : (which == 1) ? wk : wv;
    const float* bias = (which == 0) ? bq : (which == 1) ? bk : bv;
    const int tm = blockIdx.y * 128;
    const int tn = blockIdx.x * 128;
    const int wave = threadIdx.x >> 5;
    const int wm = (wave >> 1) * 32;  // 0,32,64,96
    const int wn = (wave & 1) * 64;   // 0,64
    const float* ga = x + (size_t)tm * DD;
    const float* gb = w + (size_t)tn * DD;

    v8f acc[2][4] = {};
    loadA_f32<128, 64>(lA[0], ga, DD);
    loadB_f32<128, 64>(lB[0], gb, DD);
    int cur = 0;
    for (int k0 = 0; k0 < DD; k0 += 64) {
        __syncthreads();
        if (k0 + 64 < DD) {
            loadA_f32<128, 64>(lA[cur ^ 1], ga + k0 + 64, DD);
            loadB_f32<128, 64>(lB[cur ^ 1], gb + k0 + 64, DD);
        }
#pragma unroll
        for (int s = 0; s < 2; ++s) {
            const __bf16* tA = lA[cur] + s * 128 * 32;
            const __bf16* tB = lB[cur] + s * 128 * 32;
            v16bf af[2], bfr[4];
            af[0] = ldsA_frag(tA, wm);
            af[1] = ldsA_frag(tA, wm + 16);
#pragma unroll
            for (int nf = 0; nf < 4; ++nf) bfr[nf] = ldsB_frag(tB, wn + nf * 16);
#pragma unroll
            for (int mf = 0; mf < 2; ++mf)
#pragma unroll
                for (int nf = 0; nf < 4; ++nf)
                    acc[mf][nf] = WMMA_BF16(af[mf], bfr[nf], acc[mf][nf]);
        }
        cur ^= 1;
    }
    const int lane = threadIdx.x & 31;
    const int half = lane >> 4;
    const int ln   = lane & 15;
#pragma unroll
    for (int mf = 0; mf < 2; ++mf)
#pragma unroll
        for (int nf = 0; nf < 4; ++nf)
#pragma unroll
            for (int j = 0; j < 8; ++j) {
                const int m = tm + wm + mf * 16 + half * 8 + j;
                const int n = tn + wn + nf * 16 + ln;
                const float v = acc[mf][nf][j] + bias[n];
                const int b = m >> 11, t = m & (TT - 1);
                const int h = n >> 6,  d = n & (DH - 1);
                if (which == 0)
                    qw[(((size_t)(b * HH + h)) * TT + t) * DH + d] = (__bf16)(v * 0.125f);
                else if (which == 1)
                    kw[(((size_t)(b * HH + h)) * TT + t) * DH + d] = (__bf16)v;
                else
                    vtw[(((size_t)(b * HH + h)) * DH + d) * TT + t] = (__bf16)v;
            }
}

// ---------------------------------------------------------------------------
// Kernel 2: scores = (Q*scale) @ K^T + attn_bias, masked. K = DH = 64 -> one
// LDS stage, one barrier, 16 wmma. f32 result straight into d_out attn region.
__global__ __launch_bounds__(256) void k_scores(
    const __bf16* __restrict__ qw, const __bf16* __restrict__ kw,
    const float* __restrict__ attn_bias, const unsigned char* __restrict__ kpm,
    float* __restrict__ attn) {
    __shared__ __align__(32) __bf16 lA[2 * 128 * 32];
    __shared__ __align__(32) __bf16 lB[2 * 128 * 32];
    const int bh = blockIdx.z;
    const int b  = bh >> 4;
    const int tq = blockIdx.y * 128;
    const int tk = blockIdx.x * 128;
    const __bf16* qs = qw + (size_t)bh * TT * DH + (size_t)tq * DH;
    const __bf16* ks = kw + (size_t)bh * TT * DH + (size_t)tk * DH;
    const int wave = threadIdx.x >> 5;
    const int wm = (wave >> 1) * 32;
    const int wn = (wave & 1) * 64;

    v8f acc[2][4] = {};
    loadA_bf16<128, 64>(lA, qs, DH);
    loadB_bf16<128, 64>(lB, ks, DH);
    __syncthreads();
#pragma unroll
    for (int s = 0; s < 2; ++s) {
        const __bf16* tA = lA + s * 128 * 32;
        const __bf16* tB = lB + s * 128 * 32;
        v16bf af[2], bfr[4];
        af[0] = ldsA_frag(tA, wm);
        af[1] = ldsA_frag(tA, wm + 16);
#pragma unroll
        for (int nf = 0; nf < 4; ++nf) bfr[nf] = ldsB_frag(tB, wn + nf * 16);
#pragma unroll
        for (int mf = 0; mf < 2; ++mf)
#pragma unroll
            for (int nf = 0; nf < 4; ++nf)
                acc[mf][nf] = WMMA_BF16(af[mf], bfr[nf], acc[mf][nf]);
    }
    float* out = attn + (size_t)bh * TT * TT;
    const int lane = threadIdx.x & 31;
    const int half = lane >> 4;
    const int ln   = lane & 15;
#pragma unroll
    for (int mf = 0; mf < 2; ++mf)
#pragma unroll
        for (int nf = 0; nf < 4; ++nf)
#pragma unroll
            for (int j = 0; j < 8; ++j) {
                const int q  = tq + wm + mf * 16 + half * 8 + j;
                const int kk = tk + wn + nf * 16 + ln;
                float v = acc[mf][nf][j] + attn_bias[(size_t)q * TT + kk];
                if (kpm[b * TT + kk]) v = -__builtin_inff();
                out[(size_t)q * TT + kk] = v;
            }
}

// ---------------------------------------------------------------------------
// Kernel 3: in-place row softmax. Row (8KB) cached in LDS: exactly one global
// read + one global write per element of the 1.07GB attn region.
__global__ __launch_bounds__(256) void k_softmax(float* __restrict__ attn) {
    __shared__ float row[TT];
    __shared__ float red[8];
    float* p = attn + (size_t)blockIdx.x * TT;
    const int t = threadIdx.x;

    float lmax = -3.4e38f;
#pragma unroll
    for (int i0 = 0; i0 < TT; i0 += 256) {
        const float v = p[i0 + t];
        row[i0 + t] = v;
        lmax = fmaxf(lmax, v);
    }
#pragma unroll
    for (int off = 16; off; off >>= 1) lmax = fmaxf(lmax, __shfl_xor(lmax, off, 32));
    if ((t & 31) == 0) red[t >> 5] = lmax;
    __syncthreads();
    float rmax = red[0];
#pragma unroll
    for (int i = 1; i < 8; ++i) rmax = fmaxf(rmax, red[i]);
    __syncthreads();

    float lsum = 0.f;
#pragma unroll
    for (int i0 = 0; i0 < TT; i0 += 256) {
        const float e = __expf(row[i0 + t] - rmax);
        row[i0 + t] = e;
        lsum += e;
    }
#pragma unroll
    for (int off = 16; off; off >>= 1) lsum += __shfl_xor(lsum, off, 32);
    if ((t & 31) == 0) red[t >> 5] = lsum;
    __syncthreads();
    float rsum = red[0];
#pragma unroll
    for (int i = 1; i < 8; ++i) rsum += red[i];
    const float inv = 1.0f / rsum;
#pragma unroll
    for (int i0 = 0; i0 < TT; i0 += 256) p[i0 + t] = row[i0 + t] * inv;
}

// ---------------------------------------------------------------------------
// Kernel 4: head output = attn @ V. A = attn f32 (->bf16), B = V^T [d][t] bf16.
// Tile 128(M) x 64(N=DH), KB=64, double-buffered. Context -> [b][t][h*DH+d].
__global__ __launch_bounds__(256) void k_av(
    const float* __restrict__ attn, const __bf16* __restrict__ vtw,
    __bf16* __restrict__ ctx) {
    __shared__ __align__(32) __bf16 lA[2][2 * 128 * 32];  // 16KB x2
    __shared__ __align__(32) __bf16 lB[2][2 * 64 * 32];   // 8KB x2
    const int bh = blockIdx.y;
    const int b = bh >> 4, h = bh & 15;
    const int tq = blockIdx.x * 128;
    const float*  as = attn + (size_t)bh * TT * TT + (size_t)tq * TT;
    const __bf16* vs = vtw + (size_t)bh * DH * TT;
    const int wave = threadIdx.x >> 5;
    const int wm = (wave >> 1) * 32;  // 4 waves cover 128 rows
    const int wn = (wave & 1) * 32;   // 2 waves cover 64 cols

    v8f acc[2][2] = {};
    loadA_f32<128, 64>(lA[0], as, TT);
    loadB_bf16<64, 64>(lB[0], vs, TT);
    int cur = 0;
    for (int k0 = 0; k0 < TT; k0 += 64) {
        __syncthreads();
        if (k0 + 64 < TT) {
            loadA_f32<128, 64>(lA[cur ^ 1], as + k0 + 64, TT);
            loadB_bf16<64, 64>(lB[cur ^ 1], vs + k0 + 64, TT);
        }
#pragma unroll
        for (int s = 0; s < 2; ++s) {
            const __bf16* tA = lA[cur] + s * 128 * 32;
            const __bf16* tB = lB[cur] + s * 64 * 32;
            v16bf af[2], bfr[2];
            af[0]  = ldsA_frag(tA, wm);
            af[1]  = ldsA_frag(tA, wm + 16);
            bfr[0] = ldsB_frag(tB, wn);
            bfr[1] = ldsB_frag(tB, wn + 16);
#pragma unroll
            for (int mf = 0; mf < 2; ++mf)
#pragma unroll
                for (int nf = 0; nf < 2; ++nf)
                    acc[mf][nf] = WMMA_BF16(af[mf], bfr[nf], acc[mf][nf]);
        }
        cur ^= 1;
    }
    const int lane = threadIdx.x & 31;
    const int half = lane >> 4;
    const int ln   = lane & 15;
#pragma unroll
    for (int mf = 0; mf < 2; ++mf)
#pragma unroll
        for (int nf = 0; nf < 2; ++nf)
#pragma unroll
            for (int j = 0; j < 8; ++j) {
                const int q = tq + wm + mf * 16 + half * 8 + j;
                const int d = wn + nf * 16 + ln;
                ctx[((size_t)b * TT + q) * DD + h * DH + d] = (__bf16)acc[mf][nf][j];
            }
}

// ---------------------------------------------------------------------------
// Kernel 5: out = ctx @ Wo^T + bo (f32 output into d_out[0 .. B*T*D)).
__global__ __launch_bounds__(256) void k_oproj(
    const __bf16* __restrict__ ctx, const float* __restrict__ wo,
    const float* __restrict__ bo, float* __restrict__ out) {
    __shared__ __align__(32) __bf16 lA[2][2 * 128 * 32];
    __shared__ __align__(32) __bf16 lB[2][2 * 128 * 32];
    const int tm = blockIdx.y * 128;
    const int tn = blockIdx.x * 128;
    const int wave = threadIdx.x >> 5;
    const int wm = (wave >> 1) * 32;
    const int wn = (wave & 1) * 64;
    const __bf16* ga = ctx + (size_t)tm * DD;
    const float*  gb = wo + (size_t)tn * DD;

    v8f acc[2][4] = {};
    loadA_bf16<128, 64>(lA[0], ga, DD);
    loadB_f32<128, 64>(lB[0], gb, DD);
    int cur = 0;
    for (int k0 = 0; k0 < DD; k0 += 64) {
        __syncthreads();
        if (k0 + 64 < DD) {
            loadA_bf16<128, 64>(lA[cur ^ 1], ga + k0 + 64, DD);
            loadB_f32<128, 64>(lB[cur ^ 1], gb + k0 + 64, DD);
        }
#pragma unroll
        for (int s = 0; s < 2; ++s) {
            const __bf16* tA = lA[cur] + s * 128 * 32;
            const __bf16* tB = lB[cur] + s * 128 * 32;
            v16bf af[2], bfr[4];
            af[0] = ldsA_frag(tA, wm);
            af[1] = ldsA_frag(tA, wm + 16);
#pragma unroll
            for (int nf = 0; nf < 4; ++nf) bfr[nf] = ldsB_frag(tB, wn + nf * 16);
#pragma unroll
            for (int mf = 0; mf < 2; ++mf)
#pragma unroll
                for (int nf = 0; nf < 4; ++nf)
                    acc[mf][nf] = WMMA_BF16(af[mf], bfr[nf], acc[mf][nf]);
        }
        cur ^= 1;
    }
    const int lane = threadIdx.x & 31;
    const int half = lane >> 4;
    const int ln   = lane & 15;
#pragma unroll
    for (int mf = 0; mf < 2; ++mf)
#pragma unroll
        for (int nf = 0; nf < 4; ++nf)
#pragma unroll
            for (int j = 0; j < 8; ++j) {
                const int m = tm + wm + mf * 16 + half * 8 + j;
                const int n = tn + wn + nf * 16 + ln;
                out[(size_t)m * DD + n] = acc[mf][nf][j] + bo[n];
            }
}

// ---------------------------------------------------------------------------
extern "C" void kernel_launch(void* const* d_in, const int* in_sizes, int n_in,
                              void* d_out, int out_size, void* d_ws, size_t ws_size,
                              hipStream_t stream) {
    const float* x         = (const float*)d_in[0];
    const float* attn_bias = (const float*)d_in[1];
    const unsigned char* kpm = (const unsigned char*)d_in[2];  // bool mask [B,T]
    const float* wq = (const float*)d_in[3];
    const float* bq = (const float*)d_in[4];
    const float* wk = (const float*)d_in[5];
    const float* bk = (const float*)d_in[6];
    const float* wv = (const float*)d_in[7];
    const float* bv = (const float*)d_in[8];
    const float* wo = (const float*)d_in[9];
    const float* bo = (const float*)d_in[10];

    float* out  = (float*)d_out;               // [B,T,D]
    float* attn = out + (size_t)BB * TT * DD;  // [B,H,T,T]

    const size_t headElems = (size_t)BB * HH * TT * DH;  // 8,388,608
    __bf16* qw  = (__bf16*)d_ws;               // Q  [b,h,t,d] (pre-scaled)
    __bf16* kw  = qw + headElems;              // K  [b,h,t,d]
    __bf16* vtw = kw + headElems;              // V^T[b,h,d,t]
    __bf16* ctx = vtw + headElems;             // merged-head context [b,t,D]
    // total scratch: 4 * 8,388,608 * 2B = 64 MB

    const dim3 blk(256);
    k_qkv   <<<dim3(DD / 128, MTOT / 128, 3), blk, 0, stream>>>(x, wq, bq, wk, bk, wv, bv, qw, kw, vtw);
    k_scores<<<dim3(TT / 128, TT / 128, BB * HH), blk, 0, stream>>>(qw, kw, attn_bias, kpm, attn);
    k_softmax<<<dim3(BB * HH * TT), blk, 0, stream>>>(attn);
    k_av    <<<dim3(TT / 128, BB * HH), blk, 0, stream>>>(attn, vtw, ctx);
    k_oproj <<<dim3(DD / 128, MTOT / 128), blk, 0, stream>>>(ctx, wo, bo, out);
}
